// KWSEncoder_80384607912335
// MI455X (gfx1250) — matrix operand order repaired and checked by reading, not compiled, not run on hardware
//
#include <hip/hip_runtime.h>
#include <hip/hip_bf16.h>

typedef __attribute__((ext_vector_type(16))) _Float16 v16h;
typedef __attribute__((ext_vector_type(8)))  _Float16 v8h;
typedef __attribute__((ext_vector_type(8)))  float    v8f;

#define TT     1008          // conv output time length (1000 + 2*7 - 7 + 1)
#define BB     32
#define MROWS  (BB*TT)       // 32256 rows for feedforward GEMMs
#define NAUD   1184          // 16 channels * 74 freq
#define NC     512
#define NCLASS 35
#define NRO    48            // padded readout width (3 N-tiles)

static __device__ __forceinline__ float clipf(float x, float lo, float hi) {
    return fminf(fmaxf(x, lo), hi);
}

union Frag { v16h v; struct { v8h lo; v8h hi; } p; };

#define WMMA_F16(A, B, C) \
    __builtin_amdgcn_wmma_f32_16x16x32_f16(false, (A), false, (B), (short)0, (C), false, false)

// ---------------------------------------------------------------------------
// Kernel 1: conv(7x7, pad time 7) + per-(b,t) LayerNorm over 1184 + leaky relu
// One block per (b,t). Output fp16 row-major (b*TT+t, 1184).
// ---------------------------------------------------------------------------
__global__ void __launch_bounds__(128)
k_conv_ln(const float* __restrict__ feats, const float* __restrict__ cw,
          const float* __restrict__ cb, const float* __restrict__ lng,
          const float* __restrict__ lnb, _Float16* __restrict__ out)
{
    __shared__ float sw[16 * 49];
    __shared__ float sf[7 * 80];
    __shared__ float red[128];
    __shared__ float stat[2];
    const int bt  = blockIdx.x;
    const int b   = bt / TT, t = bt % TT;
    const int tid = threadIdx.x;

    for (int i = tid; i < 16 * 49; i += 128) sw[i] = cw[i];
    for (int i = tid; i < 7 * 80; i += 128) {
        int kt = i / 80, f = i - kt * 80;
        int ti = t - 7 + kt;                       // padded conv: input rows t-7..t-1
        sf[i] = (ti >= 0 && ti < 1000) ? feats[((size_t)b * 1000 + ti) * 80 + f] : 0.f;
    }
    __syncthreads();

    float vals[10];
    float lsum = 0.f, lsq = 0.f;
    #pragma unroll
    for (int ii = 0; ii < 10; ++ii) {
        int o = tid + ii * 128;
        float v = 0.f;
        if (o < NAUD) {
            int f = o >> 4, c = o & 15;            // n = f*16 + c
            float acc = cb[c];
            #pragma unroll
            for (int kt = 0; kt < 7; ++kt)
                #pragma unroll
                for (int kf = 0; kf < 7; ++kf)
                    acc += sf[kt * 80 + f + kf] * sw[c * 49 + kt * 7 + kf];
            v = acc;
            lsum += v; lsq += v * v;
        }
        vals[ii] = v;
    }
    red[tid] = lsum; __syncthreads();
    for (int s = 64; s > 0; s >>= 1) { if (tid < s) red[tid] += red[tid + s]; __syncthreads(); }
    if (tid == 0) stat[0] = red[0] / (float)NAUD;
    __syncthreads();
    red[tid] = lsq; __syncthreads();
    for (int s = 64; s > 0; s >>= 1) { if (tid < s) red[tid] += red[tid + s]; __syncthreads(); }
    if (tid == 0) {
        float mu = stat[0];
        stat[1] = __frsqrt_rn(red[0] / (float)NAUD - mu * mu + 1e-5f);
    }
    __syncthreads();
    const float mu = stat[0], rs = stat[1];
    #pragma unroll
    for (int ii = 0; ii < 10; ++ii) {
        int o = tid + ii * 128;
        if (o < NAUD) {
            int f = o >> 4, c = o & 15;
            float y = (vals[ii] - mu) * rs * lng[f * 16 + c] + lnb[f * 16 + c];
            y = y > 0.f ? y : 0.01f * y;           // leaky relu
            out[(size_t)bt * NAUD + o] = (_Float16)y;
        }
    }
}

// ---------------------------------------------------------------------------
// Kernel 2: auditory LIF scan over time; in-place fp16 spikes + fp32 to d_out
// ---------------------------------------------------------------------------
__global__ void k_lif(_Float16* buf, const float* __restrict__ tauu,
                      float* __restrict__ spk_f32)
{
    int tid = blockIdx.x * blockDim.x + threadIdx.x;
    if (tid >= BB * NAUD) return;
    int b = tid / NAUD, n = tid - b * NAUD;
    float alpha = __expf(-1.f / clipf(tauu[n], 3.f, 25.f));
    float u = 0.f, s = 0.f;
    size_t base = (size_t)b * TT * NAUD + n;
    for (int t = 0; t < TT; ++t) {
        size_t idx = base + (size_t)t * NAUD;
        float wx = (float)buf[idx];
        u = alpha * (u - s) + (1.f - alpha) * wx;
        s = (u - 1.f > 0.f) ? 1.f : 0.f;
        spk_f32[idx] = s;
        buf[idx] = (_Float16)s;
    }
}

// ---------------------------------------------------------------------------
// Kernel 3: pack fp32 weight (Kdim x Nreal, optional elementwise mask) into
// WMMA B-fragment order: off = ((kt*NT+nt)*32 + lane)*16 + j
//   K = kt*32 + (lane>>4)*16 + j,  N = nt*16 + (lane&15)
// ---------------------------------------------------------------------------
__global__ void k_pack_b(const float* __restrict__ W, const float* __restrict__ mask,
                         _Float16* __restrict__ out, int Kdim, int Nreal, int NT)
{
    int idx = blockIdx.x * blockDim.x + threadIdx.x;
    int total = (Kdim >> 5) * NT * 512;
    if (idx >= total) return;
    int j    = idx & 15;
    int lane = (idx >> 4) & 31;
    int rem  = idx >> 9;
    int nt   = rem % NT;
    int kt   = rem / NT;
    int K = kt * 32 + ((lane >> 4) << 4) + j;
    int N = nt * 16 + (lane & 15);
    float v = 0.f;
    if (K < Kdim && N < Nreal) {
        v = W[(size_t)K * Nreal + N];
        if (mask) v *= mask[(size_t)K * Nreal + N];
    }
    out[idx] = (_Float16)v;
}

// ---------------------------------------------------------------------------
// Kernel 4: WMMA GEMM  C(Mrows x NT*16) = A(Mrows x Kdim) @ Bpack + bias
// One wave per 16x16 output tile. fp16 A from row-major, fp16 B pre-packed.
// ---------------------------------------------------------------------------
__global__ void __launch_bounds__(256)
k_gemm(const _Float16* __restrict__ A, const _Float16* __restrict__ Bp,
       const float* __restrict__ bias, float* __restrict__ C,
       int Mrows, int Kdim, int NT, int Nreal)
{
    const int wave = (blockIdx.x * blockDim.x + threadIdx.x) >> 5;
    const int lane = threadIdx.x & 31;
    const int totalTiles = (Mrows >> 4) * NT;
    if (wave >= totalTiles) return;               // uniform per wave
    const int mt = wave / NT;
    const int nt = wave - mt * NT;
    const int row  = (mt << 4) + (lane & 15);
    const int koff = (lane >> 4) << 4;            // 16-bit A frag half-select
    const int KT = Kdim >> 5;

    v8f acc = {};
    for (int kt = 0; kt < KT; ++kt) {
        Frag af, bf;
        const _Float16* ap = A + (size_t)row * Kdim + kt * 32 + koff;
        af.p.lo = *(const v8h*)(ap);
        af.p.hi = *(const v8h*)(ap + 8);
        const _Float16* bp = Bp + ((size_t)(kt * NT + nt) * 32 + lane) * 16;
        __builtin_prefetch(bp + (size_t)NT * 512, 0, 1);   // next k-tile of B
        bf.p.lo = *(const v8h*)(bp);
        bf.p.hi = *(const v8h*)(bp + 8);
        acc = WMMA_F16(af.v, bf.v, acc);
    }
    const int n = (nt << 4) + (lane & 15);
    const float bv = (n < Nreal) ? bias[n] : 0.f;
    const int Nout = NT << 4;
    #pragma unroll
    for (int r = 0; r < 8; ++r) {
        int rr = (mt << 4) + r + ((lane >> 4) << 3);
        C[(size_t)rr * Nout + n] = acc[r] + bv;
    }
}

// ---------------------------------------------------------------------------
// Kernel 5: batchnorm over all (b,t) rows per column; optional transpose to
// (t*32+b) row order so the recurrence reads one contiguous slab per step.
// One block per column.
// ---------------------------------------------------------------------------
__global__ void __launch_bounds__(256)
k_bn(const float* __restrict__ X, const float* __restrict__ g,
     const float* __restrict__ bb, float* __restrict__ Y, int Ns, int transpose)
{
    const int n = blockIdx.x;
    const int tid = threadIdx.x;
    __shared__ float red[256];
    __shared__ float stat[2];
    float sum = 0.f, sq = 0.f;
    for (int r = tid; r < MROWS; r += 256) {
        float x = X[(size_t)r * Ns + n];
        sum += x; sq += x * x;
    }
    red[tid] = sum; __syncthreads();
    for (int s = 128; s > 0; s >>= 1) { if (tid < s) red[tid] += red[tid + s]; __syncthreads(); }
    if (tid == 0) stat[0] = red[0] / (float)MROWS;
    __syncthreads();
    red[tid] = sq; __syncthreads();
    for (int s = 128; s > 0; s >>= 1) { if (tid < s) red[tid] += red[tid + s]; __syncthreads(); }
    if (tid == 0) {
        float mu = stat[0];
        stat[1] = __frsqrt_rn(red[0] / (float)MROWS - mu * mu + 1e-5f);
    }
    __syncthreads();
    const float mu = stat[0], rs = stat[1], gg = g[n], bv = bb[n];
    for (int r = tid; r < MROWS; r += 256) {
        float y = (X[(size_t)r * Ns + n] - mu) * rs * gg + bv;
        if (transpose) {
            int bi = r / TT, t = r - bi * TT;
            Y[((size_t)t * BB + bi) * Ns + n] = y;
        } else {
            Y[(size_t)r * Ns + n] = y;
        }
    }
}

// ---------------------------------------------------------------------------
// Kernel 6: persistent adLIF recurrence. One workgroup, 16 waves; each wave
// owns a 2(M) x 2(N) block of 16x16 tiles covering the 32x512 state. Spikes
// double-buffered in LDS (fp16); u/w/s states live in D-fragment layout VGPRs.
// Per step: rec = s_prev @ Vpack via WMMA, then elementwise adLIF update.
// ---------------------------------------------------------------------------
#define TILE_UPDATE(ACC, U, W, S, MOFF, NCOL, AL, BE, AA, BBC)                  \
    {                                                                           \
        int brow  = (MOFF) + r + rhalf;                                         \
        float cur = wxrow[(size_t)brow * NC + (NCOL)] + (ACC)[r];               \
        float wn  = (BE) * (W)[r] + (AA) * (U)[r] + (BBC) * (S)[r];             \
        float un  = (AL) * ((U)[r] - (S)[r]) + (1.f - (AL)) * (cur - wn);       \
        float sn  = (un - 1.f > 0.f) ? 1.f : 0.f;                               \
        (U)[r] = un; (W)[r] = wn; (S)[r] = sn;                                  \
        wb[(size_t)brow * NC + (NCOL)] = (_Float16)sn;                          \
        size_t gi = ((size_t)brow * TT + t) * NC + (NCOL);                      \
        s_f32[gi] = sn; s_h[gi] = (_Float16)sn;                                 \
    }

__global__ void __launch_bounds__(512)
k_adlif(const float* __restrict__ WxT, const _Float16* __restrict__ Vp,
        const float* __restrict__ tauu, const float* __restrict__ tauw,
        const float* __restrict__ apar, const float* __restrict__ bpar,
        const float* __restrict__ sfa,
        float* __restrict__ s_f32, _Float16* __restrict__ s_h)
{
    extern __shared__ _Float16 sbuf[];             // 2 * 32 * 512 fp16 = 64 KB
    const int tid  = threadIdx.x;
    const int wv   = tid >> 5, lane = tid & 31;
    const int nt0  = wv * 2, nt1 = wv * 2 + 1;
    const int nc0  = nt0 * 16 + (lane & 15);
    const int nc1  = nt1 * 16 + (lane & 15);
    const int rhalf = (lane >> 4) << 3;

    const float al0 = __expf(-1.f / clipf(tauu[nc0], 3.f, 25.f));
    const float al1 = __expf(-1.f / clipf(tauu[nc1], 3.f, 25.f));
    const float be0 = __expf(-1.f / clipf(tauw[nc0], 30.f, 350.f));
    const float be1 = __expf(-1.f / clipf(tauw[nc1], 30.f, 350.f));
    const float aa0 = clipf(apar[nc0], -0.5f, 5.f) * sfa[nc0];
    const float aa1 = clipf(apar[nc1], -0.5f, 5.f) * sfa[nc1];
    const float bc0 = clipf(bpar[nc0], 0.f, 2.f) * sfa[nc0];
    const float bc1 = clipf(bpar[nc1], 0.f, 2.f) * sfa[nc1];

    v8f u00 = {}, u01 = {}, u10 = {}, u11 = {};
    v8f w00 = {}, w01 = {}, w10 = {}, w11 = {};
    v8f s00 = {}, s01 = {}, s10 = {}, s11 = {};

    for (int i = tid; i < 2 * BB * NC; i += 512) sbuf[i] = (_Float16)0.f;
    __syncthreads();

    const int rowA = lane & 15;
    const int koff = (lane >> 4) << 4;

    for (int t = 0; t < TT; ++t) {
        const _Float16* rb = sbuf + (size_t)(t & 1) * (BB * NC);
        _Float16*       wb = sbuf + (size_t)((t + 1) & 1) * (BB * NC);

        v8f a00 = {}, a01 = {}, a10 = {}, a11 = {};
        for (int kt = 0; kt < (NC >> 5); ++kt) {
            Frag fa0, fa1, fb0, fb1;
            const int kbase = kt * 32 + koff;
            const _Float16* p0 = rb + (size_t)rowA * NC + kbase;          // M-tile 0
            const _Float16* p1 = rb + (size_t)(16 + rowA) * NC + kbase;   // M-tile 1
            fa0.p.lo = *(const v8h*)(p0);  fa0.p.hi = *(const v8h*)(p0 + 8);
            fa1.p.lo = *(const v8h*)(p1);  fa1.p.hi = *(const v8h*)(p1 + 8);
            const _Float16* q0 = Vp + ((size_t)(kt * 32 + nt0) * 32 + lane) * 16;
            const _Float16* q1 = Vp + ((size_t)(kt * 32 + nt1) * 32 + lane) * 16;
            fb0.p.lo = *(const v8h*)(q0);  fb0.p.hi = *(const v8h*)(q0 + 8);
            fb1.p.lo = *(const v8h*)(q1);  fb1.p.hi = *(const v8h*)(q1 + 8);
            a00 = WMMA_F16(fa0.v, fb0.v, a00);
            a01 = WMMA_F16(fa0.v, fb1.v, a01);
            a10 = WMMA_F16(fa1.v, fb0.v, a10);
            a11 = WMMA_F16(fa1.v, fb1.v, a11);
        }

        const float* wxrow = WxT + (size_t)t * (BB * NC);
        #pragma unroll
        for (int r = 0; r < 8; ++r) {
            TILE_UPDATE(a00, u00, w00, s00,  0, nc0, al0, be0, aa0, bc0)
            TILE_UPDATE(a01, u01, w01, s01,  0, nc1, al1, be1, aa1, bc1)
            TILE_UPDATE(a10, u10, w10, s10, 16, nc0, al0, be0, aa0, bc0)
            TILE_UPDATE(a11, u11, w11, s11, 16, nc1, al1, be1, aa1, bc1)
        }
        __syncthreads();
    }
}

// ---------------------------------------------------------------------------
// Kernel 7: leaky readout + softmax accumulation. One lane per batch element.
// ---------------------------------------------------------------------------
__global__ void k_readout(const float* __restrict__ Wx, const float* __restrict__ tauu,
                          float* __restrict__ out)
{
    int b = threadIdx.x;
    if (b >= BB) return;
    float alpha[NCLASS], u[NCLASS], acc[NCLASS];
    #pragma unroll
    for (int k = 0; k < NCLASS; ++k) {
        alpha[k] = __expf(-1.f / clipf(tauu[k], 3.f, 25.f));
        u[k] = 0.f; acc[k] = 0.f;
    }
    for (int t = 0; t < TT; ++t) {
        const float* row = Wx + ((size_t)b * TT + t) * NRO;
        float m = -1e30f;
        #pragma unroll
        for (int k = 0; k < NCLASS; ++k) {
            u[k] = alpha[k] * u[k] + (1.f - alpha[k]) * row[k];
            m = fmaxf(m, u[k]);
        }
        float se = 0.f, e[NCLASS];
        #pragma unroll
        for (int k = 0; k < NCLASS; ++k) { e[k] = __expf(u[k] - m); se += e[k]; }
        float inv = 1.f / se;
        #pragma unroll
        for (int k = 0; k < NCLASS; ++k) acc[k] += e[k] * inv;
    }
    #pragma unroll
    for (int k = 0; k < NCLASS; ++k) out[b * NCLASS + k] = acc[k];
}

// ---------------------------------------------------------------------------
// Host orchestration
// ---------------------------------------------------------------------------
extern "C" void kernel_launch(void* const* d_in, const int* in_sizes, int n_in,
                              void* d_out, int out_size, void* d_ws, size_t ws_size,
                              hipStream_t stream)
{
    (void)in_sizes; (void)n_in; (void)out_size; (void)ws_size;

    const float* feats    = (const float*)d_in[0];
    const float* conv_w   = (const float*)d_in[1];
    const float* conv_b   = (const float*)d_in[2];
    const float* ln_g     = (const float*)d_in[3];
    const float* ln_b     = (const float*)d_in[4];
    const float* aud_tauu = (const float*)d_in[5];
    const float* W_ro     = (const float*)d_in[39];
    const float* b_ro     = (const float*)d_in[40];
    const float* bng_ro   = (const float*)d_in[41];
    const float* bnb_ro   = (const float*)d_in[42];
    const float* tauu_ro  = (const float*)d_in[43];

    // workspace carve-up (256B aligned)
    char* ws = (char*)d_ws;
    size_t off = 0;
    auto carve = [&](size_t bytes) {
        size_t o = off;
        off = (off + bytes + 255) & ~(size_t)255;
        return (void*)(ws + o);
    };
    _Float16* Ah    = (_Float16*)carve((size_t)MROWS * NAUD * 2);  // activations/spikes fp16
    _Float16* Wpack = (_Float16*)carve((size_t)(NAUD / 32) * 32 * 512 * 2);
    _Float16* Vpack = (_Float16*)carve((size_t)(NC / 32) * 32 * 512 * 2);
    float*    Wx    = (float*)carve((size_t)MROWS * NC * 4);
    float*    WxT   = (float*)carve((size_t)MROWS * NC * 4);

    // output carve-up
    float* outp  = (float*)d_out;
    float* aud_o = outp + BB * NCLASS;                          // (32,1008,1184)
    float* s_o[3];
    s_o[0] = aud_o + (size_t)MROWS * NAUD;
    s_o[1] = s_o[0] + (size_t)MROWS * NC;
    s_o[2] = s_o[1] + (size_t)MROWS * NC;

    // 1) conv + layernorm + leaky relu  -> Ah (fp16)
    k_conv_ln<<<MROWS, 128, 0, stream>>>(feats, conv_w, conv_b, ln_g, ln_b, Ah);

    // 2) auditory LIF scan (in-place spikes in Ah, fp32 spikes to d_out)
    k_lif<<<(BB * NAUD) / 256, 256, 0, stream>>>(Ah, aud_tauu, aud_o);

    // 3) three adLIF cortex layers
    for (int i = 0; i < 3; ++i) {
        const float* Wi    = (const float*)d_in[6 + 11 * i + 0];
        const float* bi    = (const float*)d_in[6 + 11 * i + 1];
        const float* bngi  = (const float*)d_in[6 + 11 * i + 2];
        const float* bnbi  = (const float*)d_in[6 + 11 * i + 3];
        const float* tauui = (const float*)d_in[6 + 11 * i + 4];
        const float* tauwi = (const float*)d_in[6 + 11 * i + 5];
        const float* ai    = (const float*)d_in[6 + 11 * i + 6];
        const float* bbi   = (const float*)d_in[6 + 11 * i + 7];
        const float* Vi    = (const float*)d_in[6 + 11 * i + 8];
        const float* Vmi   = (const float*)d_in[6 + 11 * i + 9];
        const float* sfai  = (const float*)d_in[6 + 11 * i + 10];

        const int Kdim = (i == 0) ? NAUD : NC;
        const int KT   = Kdim / 32;

        // pack feedforward weights, GEMM, batchnorm (transposed to (t,b,n))
        {
            int total = KT * 32 * 512;
            k_pack_b<<<(total + 255) / 256, 256, 0, stream>>>(Wi, nullptr, Wpack, Kdim, NC, 32);
        }
        {
            int waves  = (MROWS / 16) * 32;
            int blocks = (waves * 32) / 256;
            k_gemm<<<blocks, 256, 0, stream>>>(Ah, Wpack, bi, Wx, MROWS, Kdim, 32, NC);
        }
        k_bn<<<NC, 256, 0, stream>>>(Wx, bngi, bnbi, WxT, NC, 1);

        // pack masked recurrent weights, run persistent recurrence
        {
            int total = (NC / 32) * 32 * 512;
            k_pack_b<<<(total + 255) / 256, 256, 0, stream>>>(Vi, Vmi, Vpack, NC, NC, 32);
        }
        k_adlif<<<1, 512, 2 * BB * NC * sizeof(_Float16), stream>>>(
            WxT, Vpack, tauui, tauwi, ai, bbi, sfai, s_o[i], Ah);
    }

    // 4) readout: GEMM (512 -> 35, padded to 48), batchnorm, softmax scan
    {
        int total = (NC / 32) * 3 * 512;
        k_pack_b<<<(total + 255) / 256, 256, 0, stream>>>(W_ro, nullptr, Wpack, NC, NCLASS, 3);
    }
    {
        int waves  = (MROWS / 16) * 3;
        int blocks = (waves * 32) / 256;
        k_gemm<<<blocks, 256, 0, stream>>>(Ah, Wpack, b_ro, Wx, MROWS, NC, 3, NCLASS);
    }
    k_bn<<<NCLASS, 256, 0, stream>>>(Wx, bng_ro, bnb_ro, WxT, NRO, 0);
    k_readout<<<1, 32, 0, stream>>>(WxT, tauu_ro, outp);
}
